// EquivariantUpdateLayer_36893769072774
// MI455X (gfx1250) — matrix operand order repaired and checked by reading, not compiled, not run on hardware
//
#include <hip/hip_runtime.h>

// ---------------------------------------------------------------------------
// EGNN-style equivariant update layer for MI455X (gfx1250, wave32, WMMA).
//   per edge: silu(LN(silu([h_i||h_j||d2] W1 + b1)) W2 + b2) -> silu(.W3+b3) W4 + b4
//   scale * dx scattered (atomic f32) into out = x + segment_sum(...)
// GEMMs on v_wmma_f32_16x16x32_f16; weights streamed via
// global_load_async_to_lds_b128 (ASYNCcnt ping-pong); A-fragments for the
// gather stage load straight from L2 into registers (h is L2-resident).
// ---------------------------------------------------------------------------

typedef _Float16 half8  __attribute__((ext_vector_type(8)));
typedef _Float16 v16h   __attribute__((ext_vector_type(16)));
typedef float    v8f    __attribute__((ext_vector_type(8)));
typedef float    f8v    __attribute__((ext_vector_type(8)));

#define TE    64      // edges per block
#define WSTR  40      // halves per row, weight chunk (pad: conflict-free b128)
#define MSTR  136     // halves per row, activation buffers

__device__ __forceinline__ v8f wmma16(v16h a, v16h b, v8f c) {
  return __builtin_amdgcn_wmma_f32_16x16x32_f16(
      /*neg_a=*/false, a, /*neg_b=*/false, b,
      /*c_mod=*/(short)0, c, /*reuse_a=*/false, /*reuse_b=*/false);
}

// CDNA5 async global->LDS copy, 16 bytes per lane, tracked by ASYNCcnt.
__device__ __forceinline__ void async_ld_b128(unsigned lds_off, const void* g) {
  asm volatile("global_load_async_to_lds_b128 %0, %1, off"
               :: "v"(lds_off), "v"(g) : "memory");
}
__device__ __forceinline__ void wait_async0() {
  asm volatile("s_wait_asynccnt 0x0" ::: "memory");
}

// A fragment, 16x32 f16 (M x K): built directly from global f32 h-rows.
// ISA layout: lanes 0-15 -> M=lane,    K = kb+{0..7, 16..23};
//             lanes 16-31 -> M=lane-16, K = kb+{8..15, 24..31}.
// hp already includes node row, chunk base and the per-lane k0 (0 or 8).
__device__ __forceinline__ v16h a_frag_from_h(const float* hp) {
  f8v lo = *(const f8v*)(hp);        // K = k0 .. k0+7
  f8v hi = *(const f8v*)(hp + 16);   // K = k0+16 .. k0+23
  half8 l = __builtin_convertvector(lo, half8);
  half8 h = __builtin_convertvector(hi, half8);
  return __builtin_shufflevector(l, h, 0,1,2,3,4,5,6,7,8,9,10,11,12,13,14,15);
}

// A fragment from LDS activations (f16, row stride ld halves).
__device__ __forceinline__ v16h load_a_frag(const _Float16* base, int ld,
                                            int m0, int kb, int lane) {
  int row = m0 + (lane & 15);
  int k0  = kb + ((lane >> 4) ? 8 : 0);
  const _Float16* p = base + row * ld + k0;
  half8 lo = *(const half8*)(p);
  half8 hi = *(const half8*)(p + 16);
  return __builtin_shufflevector(lo, hi, 0,1,2,3,4,5,6,7,8,9,10,11,12,13,14,15);
}

// B fragment, 32x16 f16 (K x N) from transposed weights Wt[n][k] in LDS.
// Lanes 0-15 hold K={0..7,16..23} for N=lane; lanes 16-31 hold K={8..15,24..31}.
__device__ __forceinline__ v16h load_b_frag(const _Float16* wt, int n0, int lane) {
  int col = n0 + (lane & 15);
  int k0  = (lane >> 4) ? 8 : 0;
  const _Float16* p = wt + col * WSTR + k0;
  half8 lo = *(const half8*)(p);
  half8 hi = *(const half8*)(p + 16);
  return __builtin_shufflevector(lo, hi, 0,1,2,3,4,5,6,7,8,9,10,11,12,13,14,15);
}

__device__ __forceinline__ float silu_f(float z) {
  return z / (1.0f + __expf(-z));
}

// Issue async copy of one 128x32 f16 weight K-chunk into an LDS buffer.
// 256 threads -> each thread copies 16 halves of one row half. 2 asyncs/thread.
__device__ __forceinline__ void stream_w_chunk(_Float16* dstbuf,
                                               const _Float16* wt, int ldw,
                                               int kb, int t) {
  int n = t >> 1, seg = (t & 1) * 16;
  _Float16* d = dstbuf + n * WSTR + seg;
  const _Float16* s = wt + n * ldw + kb + seg;
  async_ld_b128((unsigned)(uintptr_t)d, s);
  async_ld_b128((unsigned)(uintptr_t)(d + 8), s + 8);
}

// --------------------------- prep: f32 -> f16^T weights --------------------
__global__ void prep_weights(const float* __restrict__ W1,
                             const float* __restrict__ W2,
                             const float* __restrict__ W3,
                             _Float16* __restrict__ W1t,
                             _Float16* __restrict__ W2t,
                             _Float16* __restrict__ W3t) {
  int idx = blockIdx.x * blockDim.x + threadIdx.x;
  if (idx < 128 * 256) {                      // W1t[n][k] = W1[k][n], k<256
    int n = idx >> 8, k = idx & 255;
    W1t[n * 256 + k] = (_Float16)W1[k * 128 + n];
  } else if (idx < 128 * 256 + 128 * 128) {
    int j = idx - 128 * 256;
    int n = j >> 7, k = j & 127;
    W2t[n * 128 + k] = (_Float16)W2[k * 128 + n];
  } else if (idx < 128 * 256 + 2 * 128 * 128) {
    int j = idx - 128 * 256 - 128 * 128;
    int n = j >> 7, k = j & 127;
    W3t[n * 128 + k] = (_Float16)W3[k * 128 + n];
  }
}

// --------------------------- out = x --------------------------------------
__global__ void init_out(const float* __restrict__ x, float* __restrict__ out,
                         int n3) {
  int i = blockIdx.x * blockDim.x + threadIdx.x;
  if (i < n3) out[i] = x[i];
}

// --------------------------- main edge kernel -----------------------------
__global__ __launch_bounds__(256)
void edge_kernel(const float* __restrict__ h,  const int* __restrict__ e,
                 const float* __restrict__ dx, const float* __restrict__ d2,
                 const float* __restrict__ W1, const float* __restrict__ b1,
                 const float* __restrict__ g1, const float* __restrict__ beta1,
                 const float* __restrict__ b2, const float* __restrict__ b3,
                 const float* __restrict__ W4, const float* __restrict__ b4,
                 const _Float16* __restrict__ W1t,
                 const _Float16* __restrict__ W2t,
                 const _Float16* __restrict__ W3t,
                 float* __restrict__ out, int E_, int N_) {
  __shared__ __align__(16) _Float16 sW[2][128 * WSTR];  // 20 KB ping-pong
  __shared__ __align__(16) _Float16 sMa[TE * MSTR];     // 17 KB activations
  __shared__ __align__(16) _Float16 sMb[TE * MSTR];     // 17 KB activations
  __shared__ float sB1[128], sG[128], sBt[128], sB2[128], sB3[128];
  __shared__ float sW4[128], sWlast[128];
  __shared__ float sd2[TE], sScale[TE];
  __shared__ int   sEi[TE], sEj[TE];

  const int t    = threadIdx.x;
  const int lane = t & 31;
  const int wv   = t >> 5;
  const int m0   = (wv & 3) * 16;     // wave's 16-edge row tile
  const int n0   = (wv >> 2) * 64;    // wave's 64-col half
  const int eb   = blockIdx.x * TE;

  // Kick off stage-1 chunk-0 weight copy before anything else.
  stream_w_chunk(sW[0], W1t, 256, 0, t);

  // ---- phase 0: indices / small vectors -----------------------------------
  if (t < TE) {
    int ge = eb + t;
    bool ok = ge < E_;
    sd2[t] = ok ? d2[ge]      : 0.0f;
    sEi[t] = ok ? e[ge]       : 0;
    sEj[t] = ok ? e[E_ + ge]  : 0;
  }
  if (t < 128) {
    sB1[t] = b1[t];  sG[t]  = g1[t];  sBt[t] = beta1[t];
    sB2[t] = b2[t];  sB3[t] = b3[t];  sW4[t] = W4[t];
    sWlast[t] = W1[256 * 128 + t];    // rank-1 row for the d2 column
  }
  __syncthreads();

  const int mrow = m0 + ((lane >> 4) << 3);   // row base for C fragments
  const int colb = n0 + (lane & 15);

  // Per-lane A-row pointers are invariant for the whole stage-1 K loop.
  const int arow = m0 + (lane & 15);
  const int k0   = (lane >> 4) ? 8 : 0;
  const float* hpi = h + (size_t)sEi[arow] * 128 + k0;   // K 0..127   (h_i)
  const float* hpj = h + (size_t)sEj[arow] * 128 + k0;   // K 128..255 (h_j)

  v8f acc[4];
  #pragma unroll
  for (int i = 0; i < 4; ++i) acc[i] = (v8f){0,0,0,0,0,0,0,0};

  // ---- stage 1: [64 x 256] x W1t, K in 8 chunks of 32 ---------------------
  // Ping-pong async weight copies overlap the WMMA of the current chunk;
  // A-fragments come straight from global h (L2 resident) with f32->f16 cvt.
  #pragma unroll
  for (int kc = 0; kc < 8; ++kc) {
    wait_async0();        // own chunk-kc asyncs done (in-order ASYNCcnt)
    __syncthreads();      // all waves' chunk-kc data visible; kc-1 reads done
    if (kc + 1 < 8) stream_w_chunk(sW[(kc + 1) & 1], W1t, 256, (kc + 1) * 32, t);
    const float* hp = (kc < 4 ? hpi : hpj) + (kc & 3) * 32;
    v16h af = a_frag_from_h(hp);
    #pragma unroll
    for (int nt = 0; nt < 4; ++nt) {
      v16h bf = load_b_frag(sW[kc & 1], n0 + nt * 16, lane);
      acc[nt] = wmma16(af, bf, acc[nt]);
    }
  }

  // epilogue 1: + b1 + d2*W1[256]  -> silu -> f16 (pre-LN) into sMb
  #pragma unroll
  for (int nt = 0; nt < 4; ++nt)
    #pragma unroll
    for (int r = 0; r < 8; ++r) {
      int m = mrow + r, c = colb + nt * 16;
      float z = acc[nt][r] + sB1[c] + sd2[m] * sWlast[c];
      sMb[m * MSTR + c] = (_Float16)silu_f(z);
    }
  __syncthreads();

  // ---- LayerNorm over 128 features (4 lanes per edge, shfl_xor reduce) ----
  {
    int m = t >> 2, p = t & 3;
    const _Float16* row = sMb + m * MSTR + p * 32;
    float s = 0.0f, sq = 0.0f;
    #pragma unroll
    for (int i = 0; i < 32; ++i) { float v = (float)row[i]; s += v; sq += v * v; }
    s  += __shfl_xor(s, 1, 32);  s  += __shfl_xor(s, 2, 32);
    sq += __shfl_xor(sq, 1, 32); sq += __shfl_xor(sq, 2, 32);
    float mu   = s * (1.0f / 128.0f);
    float var  = sq * (1.0f / 128.0f) - mu * mu;
    float rstd = rsqrtf(var + 1e-5f);
    _Float16* orow = sMa + m * MSTR + p * 32;
    #pragma unroll
    for (int i = 0; i < 32; ++i) {
      int c = p * 32 + i;
      float v = ((float)row[i] - mu) * rstd * sG[c] + sBt[c];
      orow[i] = (_Float16)v;
    }
  }
  __syncthreads();

  // ---- stage 2: sMa x W2t -> silu -> sMb ----------------------------------
  #pragma unroll
  for (int i = 0; i < 4; ++i) acc[i] = (v8f){0,0,0,0,0,0,0,0};
  stream_w_chunk(sW[0], W2t, 128, 0, t);
  #pragma unroll
  for (int kc = 0; kc < 4; ++kc) {
    wait_async0();
    __syncthreads();
    if (kc + 1 < 4) stream_w_chunk(sW[(kc + 1) & 1], W2t, 128, (kc + 1) * 32, t);
    v16h af = load_a_frag(sMa, MSTR, m0, kc * 32, lane);
    #pragma unroll
    for (int nt = 0; nt < 4; ++nt) {
      v16h bf = load_b_frag(sW[kc & 1], n0 + nt * 16, lane);
      acc[nt] = wmma16(af, bf, acc[nt]);
    }
  }
  #pragma unroll
  for (int nt = 0; nt < 4; ++nt)
    #pragma unroll
    for (int r = 0; r < 8; ++r) {
      int m = mrow + r, c = colb + nt * 16;
      sMb[m * MSTR + c] = (_Float16)silu_f(acc[nt][r] + sB2[c]);
    }
  __syncthreads();

  // ---- stage 3: sMb x W3t -> silu -> sMa ----------------------------------
  #pragma unroll
  for (int i = 0; i < 4; ++i) acc[i] = (v8f){0,0,0,0,0,0,0,0};
  stream_w_chunk(sW[0], W3t, 128, 0, t);
  #pragma unroll
  for (int kc = 0; kc < 4; ++kc) {
    wait_async0();
    __syncthreads();
    if (kc + 1 < 4) stream_w_chunk(sW[(kc + 1) & 1], W3t, 128, (kc + 1) * 32, t);
    v16h af = load_a_frag(sMb, MSTR, m0, kc * 32, lane);
    #pragma unroll
    for (int nt = 0; nt < 4; ++nt) {
      v16h bf = load_b_frag(sW[kc & 1], n0 + nt * 16, lane);
      acc[nt] = wmma16(af, bf, acc[nt]);
    }
  }
  #pragma unroll
  for (int nt = 0; nt < 4; ++nt)
    #pragma unroll
    for (int r = 0; r < 8; ++r) {
      int m = mrow + r, c = colb + nt * 16;
      sMa[m * MSTR + c] = (_Float16)silu_f(acc[nt][r] + sB3[c]);
    }
  __syncthreads();

  // ---- scale = msg3 . W4 + b4 ---------------------------------------------
  {
    int m = t >> 2, p = t & 3;
    const _Float16* row = sMa + m * MSTR + p * 32;
    float s = 0.0f;
    #pragma unroll
    for (int i = 0; i < 32; ++i) s += (float)row[i] * sW4[p * 32 + i];
    s += __shfl_xor(s, 1, 32);
    s += __shfl_xor(s, 2, 32);
    if (p == 0) sScale[m] = s + b4[0];
  }
  __syncthreads();

  // ---- scatter: out[ei] += dx * scale -------------------------------------
  if (t < TE * 3) {
    int m = t / 3, c = t - m * 3;
    int ge = eb + m;
    if (ge < E_) {
      float v = dx[(size_t)ge * 3 + c] * sScale[m];
      unsafeAtomicAdd(&out[(size_t)sEi[m] * 3 + c], v);
    }
  }
}

// --------------------------- host entry -----------------------------------
extern "C" void kernel_launch(void* const* d_in, const int* in_sizes, int n_in,
                              void* d_out, int out_size, void* d_ws, size_t ws_size,
                              hipStream_t stream) {
  const float* h     = (const float*)d_in[0];
  const float* x     = (const float*)d_in[1];
  const int*   e     = (const int*)  d_in[2];
  const float* dx    = (const float*)d_in[3];
  const float* d2    = (const float*)d_in[4];
  const float* W1    = (const float*)d_in[5];
  const float* b1    = (const float*)d_in[6];
  const float* g1    = (const float*)d_in[7];
  const float* beta1 = (const float*)d_in[8];
  const float* W2    = (const float*)d_in[9];
  const float* b2    = (const float*)d_in[10];
  const float* W3    = (const float*)d_in[11];
  const float* b3    = (const float*)d_in[12];
  const float* W4    = (const float*)d_in[13];
  const float* b4    = (const float*)d_in[14];
  float* out = (float*)d_out;

  _Float16* W1t = (_Float16*)d_ws;                               // 128*256 f16
  _Float16* W2t = (_Float16*)((char*)d_ws + 131072);             // 128*128 f16
  _Float16* W3t = (_Float16*)((char*)d_ws + 131072 + 32768);     // 128*128 f16

  const int N = in_sizes[1] / 3;   // x: [N,3]
  const int E = in_sizes[4];       // d2: [E,1]

  prep_weights<<<(98304 + 255) / 256, 256, 0, stream>>>(W1, W2, W3, W1t, W2t, W3t);
  init_out<<<(N * 3 + 255) / 256, 256, 0, stream>>>(x, out, N * 3);
  edge_kernel<<<(E + TE - 1) / TE, 256, 0, stream>>>(
      h, e, dx, d2, W1, b1, g1, beta1, b2, b3, W4, b4,
      W1t, W2t, W3t, out, E, N);
}
